// RSTG_88776974008830
// MI455X (gfx1250) — matrix operand order repaired and checked by reading, not compiled, not run on hardware
//
#include <hip/hip_runtime.h>
#include <math.h>

typedef __bf16 bf16;
typedef bf16  v16bf __attribute__((ext_vector_type(16)));
typedef bf16  v8bf  __attribute__((ext_vector_type(8)));
typedef float v8f   __attribute__((ext_vector_type(8)));

constexpr int BB = 8, TT = 64, CC = 512, NN = 9;
constexpr int SEQ    = BB * NN;            // 72 LSTM sequences
constexpr int ROWS_N = BB * NN * TT;       // 4608 node rows
constexpr int ROWS_M = BB * NN * NN * TT;  // 41472 message rows
constexpr int G4     = 4 * CC;             // 2048 gate width

// ---------------------------------------------------------------------------
// WMMA helpers (CDNA5 16x16x32 bf16 -> f32)
// ---------------------------------------------------------------------------
__device__ __forceinline__ v8f wmma_bf16(v16bf a, v16bf b, v8f c) {
  return __builtin_amdgcn_wmma_f32_16x16x32_bf16(false, a, false, b, (short)0, c,
                                                 false, false);
}
__device__ __forceinline__ v16bf join16(v8bf a0, v8bf a1) {
  return __builtin_shufflevector(a0, a1, 0, 1, 2, 3, 4, 5, 6, 7,
                                 8, 9, 10, 11, 12, 13, 14, 15);
}
__device__ __forceinline__ float sigm(float x) { return 1.f / (1.f + expf(-x)); }

// ---------------------------------------------------------------------------
// Pack f32 weight [K,N] into WMMA B-fragment order (bf16):
//   Wp[((tk*(N/16)+tn)*32 + lane)*16 + v] = W[tk*32 + (lane>>4)*8 + (v&7) + (v>>3)*16][tn*16 + (lane&15)]
// One 32B per-lane contiguous chunk == one B fragment.
// ---------------------------------------------------------------------------
__global__ void __launch_bounds__(256)
pack_b_kernel(const float* __restrict__ W, bf16* __restrict__ Wp, int K, int N)
{
  size_t idx = (size_t)blockIdx.x * 256 + threadIdx.x;
  if (idx >= (size_t)K * N) return;
  int v    = idx & 15;
  int lane = (idx >> 4) & 31;
  size_t tile = idx >> 9;
  int ntiles = N >> 4;
  int tn = (int)(tile % ntiles), tk = (int)(tile / ntiles);
  int n = tn * 16 + (lane & 15);
  int k = tk * 32 + (lane >> 4) * 8 + (v & 7) + (v >> 3) * 16;
  Wp[idx] = (bf16)W[(size_t)k * N + n];
}

// ---------------------------------------------------------------------------
// Generic GEMM: out[M,N] = act( A[M,K](bf16 row-major) @ Wp (packed) + bias )
// M%32==0, N%256==0, K%64==0.  4 waves/block; wave -> 32x64 tile (2Mx4N).
// Ping-pong software pipeline (k unrolled x2): no register rotation copies.
// ---------------------------------------------------------------------------
__global__ void __launch_bounds__(128)
gemm_bf16_kernel(const bf16* __restrict__ A, const bf16* __restrict__ Wp,
                 const float* __restrict__ bias, float* __restrict__ outF,
                 bf16* __restrict__ outH, int M, int N, int K, int relu)
{
  const int lane  = threadIdx.x & 31;
  const int wave  = threadIdx.x >> 5;
  const int mBase = blockIdx.y * 32;
  const int nBase = (blockIdx.x * 4 + wave) * 64;
  if (nBase >= N) return;
  const int half = lane >> 4;
  const int mA   = lane & 15;

  v8f acc[2][4];
#pragma unroll
  for (int mi = 0; mi < 2; ++mi)
#pragma unroll
    for (int j = 0; j < 4; ++j)
#pragma unroll
      for (int e = 0; e < 8; ++e) acc[mi][j][e] = 0.f;

  const bf16* aRow0 = A + ((size_t)mBase + mA) * K + 8 * half;
  const bf16* aRow1 = aRow0 + (size_t)16 * K;
  const int nTiles = N >> 4;
  const int nt0 = nBase >> 4;

  auto loadA = [&](const bf16* r, int k0) -> v16bf {
    v8bf x0 = *(const v8bf*)(r + k0);
    v8bf x1 = *(const v8bf*)(r + k0 + 16);
    return join16(x0, x1);
  };
  auto loadB = [&](int k0, int j) -> v16bf {
    const bf16* bp =
        Wp + ((((size_t)(k0 >> 5) * nTiles + nt0 + j) << 5 | lane) << 4);
    return *(const v16bf*)bp;
  };

  // ping buffers: hold k0; pong buffers: hold k0+32
  v16bf aX0 = loadA(aRow0, 0);
  v16bf aX1 = loadA(aRow1, 0);
  v16bf bX[4];
#pragma unroll
  for (int j = 0; j < 4; ++j) bX[j] = loadB(0, j);

  for (int k0 = 0; k0 < K; k0 += 64) {
    v16bf aY0 = loadA(aRow0, k0 + 32);          // prefetch pong
    v16bf aY1 = loadA(aRow1, k0 + 32);
    v16bf bY[4];
#pragma unroll
    for (int j = 0; j < 4; ++j) bY[j] = loadB(k0 + 32, j);
#pragma unroll
    for (int j = 0; j < 4; ++j) {               // compute ping
      acc[0][j] = wmma_bf16(aX0, bX[j], acc[0][j]);
      acc[1][j] = wmma_bf16(aX1, bX[j], acc[1][j]);
    }
    if (k0 + 64 < K) {                          // prefetch next ping
      aX0 = loadA(aRow0, k0 + 64);
      aX1 = loadA(aRow1, k0 + 64);
#pragma unroll
      for (int j = 0; j < 4; ++j) bX[j] = loadB(k0 + 64, j);
    }
#pragma unroll
    for (int j = 0; j < 4; ++j) {               // compute pong
      acc[0][j] = wmma_bf16(aY0, bY[j], acc[0][j]);
      acc[1][j] = wmma_bf16(aY1, bY[j], acc[1][j]);
    }
  }

  const int mOff = half * 8, nOff = lane & 15;  // ISA C layout
#pragma unroll
  for (int mi = 0; mi < 2; ++mi) {
#pragma unroll
    for (int j = 0; j < 4; ++j) {
      int col = nBase + 16 * j + nOff;
      float bv = bias ? bias[col] : 0.f;
#pragma unroll
      for (int e = 0; e < 8; ++e) {
        int row = mBase + mi * 16 + mOff + e;
        float v = acc[mi][j][e] + bv;
        if (relu) v = fmaxf(v, 0.f);
        size_t idx = (size_t)row * N + col;
        if (outF) outF[idx] = v;
        if (outH) outH[idx] = (bf16)v;
      }
    }
  }
}

// ---------------------------------------------------------------------------
// LSTM recurrent kernel: 16 sequences per block, 8 waves.
// Wave w owns hidden tiles {w,w+8,w+16,w+24}; computes i,f,g,o gate tiles with
// WMMA over K=512 (ping-pong pipelined), cell update in-register; h via LDS.
// WhhP is in packed B-fragment order (K=512, N=2048).
// ---------------------------------------------------------------------------
__global__ void __launch_bounds__(256)
lstm_rec_kernel(const float* __restrict__ xproj,   // [(b*N+n), T, 2048]
                const bf16* __restrict__ WhhP,     // packed [512,2048]
                const float* __restrict__ bih, const float* __restrict__ bhh,
                float* __restrict__ nodesF, bf16* __restrict__ nodesH)
{
  __shared__ bf16 hlds[16][544];                    // 544: keep 16B alignment
  const int lane = threadIdx.x & 31;
  const int wave = threadIdx.x >> 5;
  const int seq0 = blockIdx.x * 16;
  const int half = lane >> 4;
  const int mA   = lane & 15;      // A-frag row
  const int mC   = half * 8;       // C-layout row offset (add e)
  const int nC   = lane & 15;      // C-layout col

  for (int i = threadIdx.x; i < 16 * 544; i += blockDim.x)
    (&hlds[0][0])[i] = (bf16)0.f;

  v8f cstate[4];
#pragma unroll
  for (int u = 0; u < 4; ++u)
#pragma unroll
    for (int e = 0; e < 8; ++e) cstate[u][e] = 0.f;
  __syncthreads();

  auto loadHA = [&](int kk) -> v16bf {
    const bf16* hp = &hlds[mA][kk * 32 + 8 * half];
    return join16(*(const v8bf*)hp, *(const v8bf*)(hp + 16));
  };

  float hnew[4][8];
  for (int t = 0; t < TT; ++t) {
    // ---- compute phase (reads old h from LDS) ----
    for (int u = 0; u < 4; ++u) {
      const int ut = wave + u * 8;                 // hidden tile 0..31
      auto loadWB = [&](int kk, int g) -> v16bf {
        return *(const v16bf*)(WhhP +
            ((((size_t)kk * 128 + g * 32 + ut) << 5 | lane) << 4));
      };
      v8f acc[4];
#pragma unroll
      for (int g = 0; g < 4; ++g) {
        const int col = g * CC + ut * 16 + nC;
        const float bsum = bih[col] + bhh[col];
#pragma unroll
        for (int e = 0; e < 8; ++e) {
          int s = seq0 + mC + e;
          float xv = (s < SEQ) ? xproj[((size_t)s * TT + t) * G4 + col] : 0.f;
          acc[g][e] = xv + bsum;
        }
      }
      // ping-pong over kk (16 steps, unrolled x2)
      v16bf aX = loadHA(0);
      v16bf bX[4];
#pragma unroll
      for (int g = 0; g < 4; ++g) bX[g] = loadWB(0, g);
      for (int kk = 0; kk < 16; kk += 2) {
        v16bf aY = loadHA(kk + 1);                 // prefetch pong
        v16bf bY[4];
#pragma unroll
        for (int g = 0; g < 4; ++g) bY[g] = loadWB(kk + 1, g);
#pragma unroll
        for (int g = 0; g < 4; ++g) acc[g] = wmma_bf16(aX, bX[g], acc[g]);
        if (kk + 2 < 16) {                         // prefetch next ping
          aX = loadHA(kk + 2);
#pragma unroll
          for (int g = 0; g < 4; ++g) bX[g] = loadWB(kk + 2, g);
        }
#pragma unroll
        for (int g = 0; g < 4; ++g) acc[g] = wmma_bf16(aY, bY[g], acc[g]);
      }
#pragma unroll
      for (int e = 0; e < 8; ++e) {                // LSTM cell, torch gate order
        float iv = sigm(acc[0][e]);
        float fv = sigm(acc[1][e]);
        float gv = tanhf(acc[2][e]);
        float ov = sigm(acc[3][e]);
        float c  = fv * cstate[u][e] + iv * gv;
        cstate[u][e] = c;
        hnew[u][e]   = ov * tanhf(c);
      }
    }
    __syncthreads();                               // everyone done reading h
    // ---- write phase ----
    for (int u = 0; u < 4; ++u) {
      const int ut = wave + u * 8;
#pragma unroll
      for (int e = 0; e < 8; ++e) {
        int m = mC + e, col = ut * 16 + nC;
        hlds[m][col] = (bf16)hnew[u][e];
        int s = seq0 + m;
        if (s < SEQ) {
          size_t idx = ((size_t)s * TT + t) * CC + col;
          nodesF[idx] = hnew[u][e];
          nodesH[idx] = (bf16)hnew[u][e];
        }
      }
    }
    __syncthreads();
  }
}

// ---------------------------------------------------------------------------
// Block reduction (wave32 shuffles + 8-wave combine)
// ---------------------------------------------------------------------------
__device__ __forceinline__ float block_sum(float v, float* s8) {
  const int lane = threadIdx.x & 31, w = threadIdx.x >> 5;
#pragma unroll
  for (int off = 16; off > 0; off >>= 1) v += __shfl_down(v, off, 32);
  if (lane == 0) s8[w] = v;
  __syncthreads();
  if (threadIdx.x == 0) {
    float s = 0.f;
#pragma unroll
    for (int i = 0; i < 8; ++i) s += s8[i];
    s8[0] = s;
  }
  __syncthreads();
  return s8[0];
}

// Transpose [B,T,C,N] -> [B,N,T,C] fused with LayerNorm(gamma_bg[n],beta_bg[n])
__global__ void __launch_bounds__(256)
prep_kernel(const float* __restrict__ in, const float* __restrict__ gamma,
            const float* __restrict__ beta, float* __restrict__ nodesF,
            bf16* __restrict__ nodesH)
{
  __shared__ float s1[8], s2[8];
  int row = blockIdx.x;                       // (b,n,t)
  int t = row % TT, n = (row / TT) % NN, b = row / (TT * NN);
  const float* base = in + ((size_t)(b * TT + t) * CC) * NN + n;
  int c0 = threadIdx.x, c1 = threadIdx.x + 256;
  float v0 = base[(size_t)c0 * NN];
  float v1 = base[(size_t)c1 * NN];
  float sum = block_sum(v0 + v1, s1);
  float sq  = block_sum(v0 * v0 + v1 * v1, s2);
  float mu = sum * (1.f / CC);
  float rs = rsqrtf(sq * (1.f / CC) - mu * mu + 1e-5f);
  const float* g  = gamma + (size_t)n * CC;
  const float* be = beta  + (size_t)n * CC;
  float y0 = (v0 - mu) * rs * g[c0] + be[c0];
  float y1 = (v1 - mu) * rs * g[c1] + be[c1];
  size_t o = (size_t)row * CC;
  nodesF[o + c0] = y0;        nodesF[o + c1] = y1;
  nodesH[o + c0] = (bf16)y0;  nodesH[o + c1] = (bf16)y1;
}

// Generic row LayerNorm, affine index = (row/T) % A
__global__ void __launch_bounds__(256)
ln_rows_kernel(const float* __restrict__ x, const float* __restrict__ gamma,
               const float* __restrict__ beta, float* __restrict__ outF,
               bf16* __restrict__ outH, int A)
{
  __shared__ float s1[8], s2[8];
  int row = blockIdx.x;
  const float* xr = x + (size_t)row * CC;
  int c0 = threadIdx.x, c1 = threadIdx.x + 256;
  float v0 = xr[c0], v1 = xr[c1];
  float sum = block_sum(v0 + v1, s1);
  float sq  = block_sum(v0 * v0 + v1 * v1, s2);
  float mu = sum * (1.f / CC);
  float rs = rsqrtf(sq * (1.f / CC) - mu * mu + 1e-5f);
  int aidx = (row / TT) % A;
  const float* g  = gamma + (size_t)aidx * CC;
  const float* be = beta  + (size_t)aidx * CC;
  float y0 = (v0 - mu) * rs * g[c0] + be[c0];
  float y1 = (v1 - mu) * rs * g[c1] + be[c1];
  size_t o = (size_t)row * CC;
  if (outF) { outF[o + c0] = y0; outF[o + c1] = y1; }
  if (outH) { outH[o + c0] = (bf16)y0; outH[o + c1] = (bf16)y1; }
}

// m1[(b,i,j,t),c] = relu(P[(b,j,t),c] + Q[(b,i,t),c] + b1[c])  (concat-GEMM split)
__global__ void __launch_bounds__(256)
combine_m1_kernel(const float* __restrict__ P, const float* __restrict__ Q,
                  const float* __restrict__ b1, bf16* __restrict__ m1)
{
  int idx = blockIdx.x * 256 + threadIdx.x;
  if (idx >= ROWS_M * CC) return;
  int c = idx & (CC - 1);
  int row = idx >> 9;
  int t = row % TT, j = (row / TT) % NN, i = (row / (TT * NN)) % NN,
      b = row / (TT * NN * NN);
  float v = P[(((size_t)(b * NN + j)) * TT + t) * CC + c] +
            Q[(((size_t)(b * NN + i)) * TT + t) * CC + c] + b1[c];
  m1[idx] = (bf16)fmaxf(v, 0.f);
}

// u1[(b,n,t),c] = relu(R + S + b1[c])
__global__ void __launch_bounds__(256)
combine_u1_kernel(const float* __restrict__ R, const float* __restrict__ S,
                  const float* __restrict__ b1, bf16* __restrict__ u1)
{
  int idx = blockIdx.x * 256 + threadIdx.x;
  if (idx >= ROWS_N * CC) return;
  int c = idx & (CC - 1);
  u1[idx] = (bf16)fmaxf(R[idx] + S[idx] + b1[c], 0.f);
}

// Attention: block per (b,t); 81 dots of length 512, softmax over j.
__global__ void __launch_bounds__(256)
attn_kernel(const float* __restrict__ q, const float* __restrict__ k,
            float* __restrict__ att)
{
  __shared__ float qs[NN * CC], ks[NN * CC], corr[NN * NN];
  int bt = blockIdx.x, b = bt / TT, t = bt % TT;
  for (int idx = threadIdx.x; idx < NN * CC; idx += 256) {
    int i = idx >> 9, c = idx & (CC - 1);
    size_t g = (((size_t)(b * NN + i)) * TT + t) * CC + c;
    qs[idx] = q[g];
    ks[idx] = k[g];
  }
  __syncthreads();
  int lane = threadIdx.x & 31, wave = threadIdx.x >> 5;
  for (int d = wave; d < NN * NN; d += 8) {
    int i = d / NN, j = d % NN;
    float p = 0.f;
    for (int c = lane; c < CC; c += 32) p += qs[i * CC + c] * ks[j * CC + c];
#pragma unroll
    for (int off = 16; off > 0; off >>= 1) p += __shfl_down(p, off, 32);
    if (lane == 0) corr[d] = p;
  }
  __syncthreads();
  if (threadIdx.x < NN) {
    int i = threadIdx.x;
    float mx = -1e30f;
#pragma unroll
    for (int j = 0; j < NN; ++j) mx = fmaxf(mx, corr[i * NN + j]);
    float e[NN], s = 0.f;
#pragma unroll
    for (int j = 0; j < NN; ++j) { e[j] = expf(corr[i * NN + j] - mx); s += e[j]; }
    float inv = 1.f / s;
#pragma unroll
    for (int j = 0; j < NN; ++j)
      att[(((size_t)(b * NN + i)) * NN + j) * TT + t] = e[j] * inv;
  }
}

// agg[(b,i,t),c] = relu(sum_j att * m + att_bias[c])
__global__ void __launch_bounds__(256)
agg_kernel(const float* __restrict__ att, const bf16* __restrict__ m,
           const float* __restrict__ abias, bf16* __restrict__ agg)
{
  int idx = blockIdx.x * 256 + threadIdx.x;
  if (idx >= ROWS_N * CC) return;
  int c = idx & (CC - 1);
  int row = idx >> 9;
  int t = row % TT, i = (row / TT) % NN, b = row / (TT * NN);
  float s = 0.f;
#pragma unroll
  for (int j = 0; j < NN; ++j) {
    size_t mr = (((size_t)(b * NN + i)) * NN + j) * TT + t;
    s += att[mr] * (float)m[mr * CC + c];
  }
  agg[idx] = (bf16)fmaxf(s + abias[c], 0.f);
}

// [B,N,T,C] -> [B,T,C,N]
__global__ void __launch_bounds__(256)
out_kernel(const float* __restrict__ nodesF, float* __restrict__ out)
{
  int idx = blockIdx.x * 256 + threadIdx.x;
  if (idx >= BB * TT * CC * NN) return;
  int n = idx % NN;
  int r = idx / NN;
  int c = r % CC; r /= CC;
  int t = r % TT; int b = r / TT;
  out[idx] = nodesF[(((size_t)(b * NN + n)) * TT + t) * CC + c];
}

// ---------------------------------------------------------------------------
extern "C" void kernel_launch(void* const* d_in, const int* in_sizes, int n_in,
                              void* d_out, int out_size, void* d_ws, size_t ws_size,
                              hipStream_t stream)
{
  (void)in_sizes; (void)n_in; (void)out_size; (void)ws_size;
  const float* input    = (const float*)d_in[0];
  const float* gamma_bg = (const float*)d_in[1];
  const float* beta_bg  = (const float*)d_in[2];
  const float* gamma_sm = (const float*)d_in[3];
  const float* beta_sm  = (const float*)d_in[4];
  const float* gamma_up = (const float*)d_in[5];
  const float* beta_up  = (const float*)d_in[6];
  const float* send_W1  = (const float*)d_in[7];
  const float* send_b1  = (const float*)d_in[8];
  const float* send_W2  = (const float*)d_in[9];
  const float* send_b2  = (const float*)d_in[10];
  const float* Wq       = (const float*)d_in[11];
  const float* bq       = (const float*)d_in[12];
  const float* Wk       = (const float*)d_in[13];
  const float* bk       = (const float*)d_in[14];
  const float* att_bias = (const float*)d_in[15];
  const float* upd_W1   = (const float*)d_in[16];
  const float* upd_b1   = (const float*)d_in[17];
  const float* upd_W2   = (const float*)d_in[18];
  const float* upd_b2   = (const float*)d_in[19];
  const float* iWih = (const float*)d_in[20];
  const float* iWhh = (const float*)d_in[21];
  const float* ibih = (const float*)d_in[22];
  const float* ibhh = (const float*)d_in[23];
  const float* eWih = (const float*)d_in[24];
  const float* eWhh = (const float*)d_in[25];
  const float* ebih = (const float*)d_in[26];
  const float* ebhh = (const float*)d_in[27];

  char* p = (char*)d_ws;
  size_t off = 0;
  auto alloc = [&](size_t bytes) -> void* {
    void* r = (void*)(p + off);
    off += (bytes + 255) & ~(size_t)255;
    return r;
  };
  float* nodesF = (float*)alloc((size_t)ROWS_N * CC * 4);
  bf16*  nodesH = (bf16*) alloc((size_t)ROWS_N * CC * 2);
  bf16*  wihI = (bf16*)alloc((size_t)CC * G4 * 2);
  bf16*  whhI = (bf16*)alloc((size_t)CC * G4 * 2);
  bf16*  wihE = (bf16*)alloc((size_t)CC * G4 * 2);
  bf16*  whhE = (bf16*)alloc((size_t)CC * G4 * 2);
  bf16*  sW1t = (bf16*)alloc((size_t)CC * CC * 2);
  bf16*  sW1b = (bf16*)alloc((size_t)CC * CC * 2);
  bf16*  sW2P = (bf16*)alloc((size_t)CC * CC * 2);
  bf16*  wqP  = (bf16*)alloc((size_t)CC * CC * 2);
  bf16*  wkP  = (bf16*)alloc((size_t)CC * CC * 2);
  bf16*  uW1t = (bf16*)alloc((size_t)CC * CC * 2);
  bf16*  uW1b = (bf16*)alloc((size_t)CC * CC * 2);
  bf16*  uW2P = (bf16*)alloc((size_t)CC * CC * 2);
  float* xproj = (float*)alloc((size_t)ROWS_N * G4 * 4);
  float* Pf = (float*)alloc((size_t)ROWS_N * CC * 4);   // reused as R
  float* Qf = (float*)alloc((size_t)ROWS_N * CC * 4);   // reused as S
  bf16*  m1H = (bf16*)alloc((size_t)ROWS_M * CC * 2);   // reused as u1
  float* m2F = (float*)alloc((size_t)ROWS_M * CC * 4);  // reused as u2
  bf16*  mH  = (bf16*) alloc((size_t)ROWS_M * CC * 2);
  float* qF  = (float*)alloc((size_t)ROWS_N * CC * 4);
  float* kF  = (float*)alloc((size_t)ROWS_N * CC * 4);
  float* attF = (float*)alloc((size_t)(BB * NN * NN * TT) * 4);
  bf16*  aggH = (bf16*)alloc((size_t)ROWS_N * CC * 2);

  auto pack = [&](const float* s, bf16* d, int K_, int N_) {
    size_t n = (size_t)K_ * N_;
    pack_b_kernel<<<(unsigned)((n + 255) / 256), 256, 0, stream>>>(s, d, K_, N_);
  };
  pack(iWih, wihI, CC, G4);  pack(iWhh, whhI, CC, G4);
  pack(eWih, wihE, CC, G4);  pack(eWhh, whhE, CC, G4);
  pack(send_W1, sW1t, CC, CC);                 // top half: rows 0..C-1
  pack(send_W1 + (size_t)CC * CC, sW1b, CC, CC);
  pack(send_W2, sW2P, CC, CC);
  pack(Wq, wqP, CC, CC);     pack(Wk, wkP, CC, CC);
  pack(upd_W1, uW1t, CC, CC);
  pack(upd_W1 + (size_t)CC * CC, uW1b, CC, CC);
  pack(upd_W2, uW2P, CC, CC);

  auto gemm = [&](const bf16* A, const bf16* W, const float* bias, float* oF,
                  bf16* oH, int M, int N, int K, int relu) {
    dim3 grid(N / 256, M / 32);
    gemm_bf16_kernel<<<grid, 128, 0, stream>>>(A, W, bias, oF, oH, M, N, K, relu);
  };

  prep_kernel<<<ROWS_N, 256, 0, stream>>>(input, gamma_bg, beta_bg, nodesF, nodesH);

  for (int iter = 0; iter < 3; ++iter) {
    // internal LSTM: x@Wih hoisted, then recurrence (16 seqs/block, WMMA M=16)
    gemm(nodesH, wihI, nullptr, xproj, nullptr, ROWS_N, G4, CC, 0);
    lstm_rec_kernel<<<(SEQ + 15) / 16, 256, 0, stream>>>(xproj, whhI, ibih, ibhh,
                                                         nodesF, nodesH);
    // send messages: cat-GEMM split  P = n@W1_top (j-term), Q = n@W1_bot (i-term)
    gemm(nodesH, sW1t, nullptr, Pf, nullptr, ROWS_N, CC, CC, 0);
    gemm(nodesH, sW1b, nullptr, Qf, nullptr, ROWS_N, CC, CC, 0);
    combine_m1_kernel<<<((size_t)ROWS_M * CC + 255) / 256, 256, 0, stream>>>(
        Pf, Qf, send_b1, m1H);
    gemm(m1H, sW2P, send_b2, m2F, nullptr, ROWS_M, CC, CC, 1);
    ln_rows_kernel<<<ROWS_M, 256, 0, stream>>>(m2F, gamma_sm, beta_sm, nullptr,
                                               mH, NN * NN);
    // attention
    gemm(nodesH, wqP, bq, qF, nullptr, ROWS_N, CC, CC, 0);
    gemm(nodesH, wkP, bk, kF, nullptr, ROWS_N, CC, CC, 0);
    attn_kernel<<<BB * TT, 256, 0, stream>>>(qF, kF, attF);
    agg_kernel<<<((size_t)ROWS_N * CC + 255) / 256, 256, 0, stream>>>(
        attF, mH, att_bias, aggH);
    // update: cat-GEMM split again
    gemm(nodesH, uW1t, nullptr, Pf, nullptr, ROWS_N, CC, CC, 0);
    gemm(aggH, uW1b, nullptr, Qf, nullptr, ROWS_N, CC, CC, 0);
    combine_u1_kernel<<<((size_t)ROWS_N * CC + 255) / 256, 256, 0, stream>>>(
        Pf, Qf, upd_b1, m1H);
    gemm(m1H, uW2P, upd_b2, m2F, nullptr, ROWS_N, CC, CC, 1);
    ln_rows_kernel<<<ROWS_N, 256, 0, stream>>>(m2F, gamma_up, beta_up, nodesF,
                                               nodesH, NN);
  }

  // external LSTM
  gemm(nodesH, wihE, nullptr, xproj, nullptr, ROWS_N, G4, CC, 0);
  lstm_rec_kernel<<<(SEQ + 15) / 16, 256, 0, stream>>>(xproj, whhE, ebih, ebhh,
                                                       nodesF, nodesH);

  out_kernel<<<(BB * TT * CC * NN + 255) / 256, 256, 0, stream>>>(nodesF,
                                                                  (float*)d_out);
}
